// YAggregator_5909874999599
// MI455X (gfx1250) — compile-verified, matrix operand order checked by reading, not run on hardware
//
#include <hip/hip_runtime.h>
#include <hip/hip_bf16.h>
#include <math.h>

typedef unsigned short u16;
typedef unsigned int   u32;
typedef unsigned long long u64;
typedef __attribute__((ext_vector_type(16))) __bf16 v16bf;
typedef __attribute__((ext_vector_type(8)))  float  v8f;

// ---------- helpers ----------
__device__ __forceinline__ u16 f2bf(float f) {
    u32 u = __builtin_bit_cast(u32, f);
    u32 r = (u + 0x7FFFu + ((u >> 16) & 1u)) >> 16;   // round-to-nearest-even
    return (u16)r;
}

// Load one wave32 WMMA bf16 fragment (A or B, mirror layouts) from a
// k-contiguous row.  Lane half selects which k-chunks this lane holds:
// v0-3 <- k[half*8 .. +7], v4-7 <- k[16+half*8 .. +7].  Two b128 loads.
__device__ __forceinline__ v16bf load_frag(const u16* base, int half) {
    union { uint4 q[2]; v16bf v; } u;
    u.q[0] = *(const uint4*)(base + half * 8);
    u.q[1] = *(const uint4*)(base + 16 + half * 8);
    return u.v;
}

__device__ __forceinline__ v8f wmma_bf16(v16bf a, v16bf b, v8f c) {
    return __builtin_amdgcn_wmma_f32_16x16x32_bf16(
        false, a, false, b, (short)0, c, false, false);
}

__device__ __forceinline__ float wred32(float s) {
#pragma unroll
    for (int o = 16; o > 0; o >>= 1) s += __shfl_xor(s, o, 32);
    return s;
}

// ---------- constants ----------
#define KM   128            // MSA depth
#define LL   256            // sequence length
#define NINP 384
#define HC   64             // nh*nhid
#define NH   4
#define CC   16             // nhid
#define MCOL  (LL*CC)       // 4096 per-head gram dim
#define PROJ_IN 1152
#define AGG   256

// ---------- prep: fp32 -> bf16 conversions ----------
__global__ __launch_bounds__(256) void k_cvt_x1d(const float* __restrict__ x, u16* __restrict__ o) {
    size_t i = (size_t)blockIdx.x * 2048 + threadIdx.x;
#pragma unroll
    for (int t = 0; t < 8; ++t, i += 256)
        o[i] = f2bf(x[i]);
}

__global__ __launch_bounds__(256) void k_cvt_w(const float* __restrict__ Wp, const float* __restrict__ Wq,
                                               u16* __restrict__ wt) {
    int e = blockIdx.x * 256 + threadIdx.x;      // 128*384 = 49152
    if (e >= 128 * NINP) return;
    int n = e / NINP, k = e - n * NINP;
    float v = (n < HC) ? Wp[(size_t)k * HC + n] : Wq[(size_t)k * HC + (n - HC)];
    wt[e] = f2bf(v);                             // wt[n][k], k contiguous
}

__global__ __launch_bounds__(256) void k_cvt_wo(const float* __restrict__ Wo, u16* __restrict__ wot) {
    int e = blockIdx.x * 256 + threadIdx.x;      // 256*1152 = 294912
    int n = e / PROJ_IN, k = e - n * PROJ_IN;
    wot[e] = f2bf(Wo[(size_t)k * AGG + n]);      // wot[n][k], k contiguous
}

// ---------- kernel 1: fused P/Q projection GEMM (32768 x 128, K=384) ----------
__global__ __launch_bounds__(256) void k_gemm_pq(const u16* __restrict__ xb, const u16* __restrict__ wt,
                                                 const float* __restrict__ bp, const float* __restrict__ bq,
                                                 float* __restrict__ Praw, float* __restrict__ Qraw) {
    int wave = threadIdx.x >> 5, lane = threadIdx.x & 31;
    int half = lane >> 4, l16 = lane & 15;
    int mbase = blockIdx.x * 128 + wave * 16;
    const u16* arow = xb + (size_t)(mbase + l16) * NINP;
    const u16* brow = wt + (size_t)l16 * NINP;   // + nt*16*NINP per n-tile

    v8f acc[8];
#pragma unroll
    for (int n = 0; n < 8; ++n) acc[n] = (v8f)(0.f);

    v16bf a = load_frag(arow, half);             // software-pipelined A
    for (int kt = 0; kt < NINP / 32; ++kt) {
        int k0 = kt * 32;
        v16bf b[8];
#pragma unroll
        for (int nt = 0; nt < 8; ++nt)
            b[nt] = load_frag(brow + (size_t)nt * (16 * NINP) + k0, half);
        v16bf an = (kt + 1 < NINP / 32) ? load_frag(arow + k0 + 32, half) : a;
#pragma unroll
        for (int nt = 0; nt < 8; ++nt)
            acc[nt] = wmma_bf16(a, b[nt], acc[nt]);
        a = an;
    }
#pragma unroll
    for (int nt = 0; nt < 8; ++nt) {
        int col = nt * 16 + l16;
#pragma unroll
        for (int v = 0; v < 8; ++v) {
            int row = mbase + v + half * 8;
            float val = acc[nt][v];
            if (col < HC) {
                Praw[(size_t)row * HC + col] = val + bp[col];
            } else {
                float q = val + bq[col - HC];
                q = fminf(fmaxf(q, -20.f), 20.f);
                Qraw[(size_t)row * HC + (col - HC)] = q;
            }
        }
    }
}

// ---------- kernel 2a: max over K axis for Q ----------
__global__ __launch_bounds__(256) void k_colmax(const float* __restrict__ Qraw, float* __restrict__ cmax) {
    int idx = blockIdx.x * 256 + threadIdx.x;    // idx = l*64 + ch, 16384 total
    float m = -1e30f;
    for (int k = 0; k < KM; ++k)
        m = fmaxf(m, Qraw[(size_t)k * (LL * HC) + idx]);
    cmax[idx] = m;
}

// ---------- kernel 2b: LN(P), exp(Q-max), pack bf16 column-major per head ----------
__global__ __launch_bounds__(256) void k_lnexp(const float* __restrict__ Praw, const float* __restrict__ Qraw,
                                               const float* __restrict__ cmax,
                                               const float* __restrict__ gp, const float* __restrict__ betap,
                                               u16* __restrict__ PQT, u16* __restrict__ QT) {
    int wave = threadIdx.x >> 5, lane = threadIdx.x & 31;
    int row = blockIdx.x * 8 + wave;             // row = k*256 + l
    int k = row >> 8, l = row & 255;
    int ch0 = lane, ch1 = lane + 32;

    float p0 = Praw[(size_t)row * HC + ch0];
    float p1 = Praw[(size_t)row * HC + ch1];
    float mean = wred32(p0 + p1) * (1.f / 64.f);
    float d0 = p0 - mean, d1 = p1 - mean;
    float var = wred32(d0 * d0 + d1 * d1) * (1.f / 64.f);
    float inv = rsqrtf(var + 1e-5f);
    float pl0 = d0 * inv * gp[ch0] + betap[ch0];
    float pl1 = d1 * inv * gp[ch1] + betap[ch1];

    float q0 = expf(Qraw[(size_t)row * HC + ch0] - cmax[l * HC + ch0]);
    float q1 = expf(Qraw[(size_t)row * HC + ch1] - cmax[l * HC + ch1]);

#pragma unroll
    for (int s = 0; s < 2; ++s) {
        int   ch = s ? ch1 : ch0;
        float pq = s ? (pl1 * q1) : (pl0 * q0);
        float qq = s ? q1 : q0;
        int h = ch >> 4, a = ch & 15;
        size_t idx = ((size_t)h * MCOL + (l * CC + a)) * KM + k;   // [h][m][k]
        PQT[idx] = f2bf(pq);
        QT [idx] = f2bf(qq);
    }
}

// ---------- kernel 3: fused Gram(X,Y) -> ratio -> concat -> out GEMM -> LN -> ReLU ----------
__global__ __launch_bounds__(256) void k_fused(const u16* __restrict__ PQT, const u16* __restrict__ QT,
                                               const float* __restrict__ x2d, const u16* __restrict__ wot,
                                               const float* __restrict__ bo, const float* __restrict__ go,
                                               const float* __restrict__ betao, float* __restrict__ out) {
    __shared__ u16   agg[16][PROJ_IN];    // 36 KB, bf16 agg rows for 16 (i,j) pairs
    __shared__ float otile[16][AGG];      // 16 KB, fp32 out tile pre-LN
    __shared__ float x2ds[16][128];       //  8 KB, async-staged x2d tile

    int i  = blockIdx.y;                  // 0..255
    int jb = blockIdx.x;                  // 0..15, j = jb*16 + r
    int tid = threadIdx.x, wave = tid >> 5, lane = tid & 31;
    int half = lane >> 4, l16 = lane & 15;

    // ---- kick off async DMA of the x2d tile into LDS (overlaps Gram WMMAs) ----
#pragma unroll
    for (int it = 0; it < 2; ++it) {
        int e  = tid + it * 256;          // 512 float4 packets
        int r  = e >> 5, c4 = (e & 31) * 4;
        u64 gsrc = (u64)(x2d + (((size_t)i * LL) + (jb * 16 + r)) * 128 + c4);
        u32 loff = (u32)(size_t)&x2ds[r][c4];
        asm volatile("global_load_async_to_lds_b128 %0, %1, off"
                     :: "v"(loff), "v"(gsrc) : "memory");
    }

    // ---- stage 1: per-(h,j) 16x16x128 Gram blocks via WMMA, fused ratio ----
    int h   = wave >> 1;
    int jlo = (wave & 1) * 8;
    const u16* Apq = PQT + ((size_t)h * MCOL + i * CC + l16) * KM;
    const u16* Aq  = QT  + ((size_t)h * MCOL + i * CC + l16) * KM;
    v16bf apq[4], aq[4];
#pragma unroll
    for (int kt = 0; kt < 4; ++kt) {
        apq[kt] = load_frag(Apq + kt * 32, half);
        aq [kt] = load_frag(Aq  + kt * 32, half);
    }
    for (int jj = 0; jj < 8; ++jj) {
        int r = jlo + jj;                 // agg row within tile
        int j = jb * 16 + r;
        const u16* Bpq = PQT + ((size_t)h * MCOL + j * CC + l16) * KM;
        const u16* Bq  = QT  + ((size_t)h * MCOL + j * CC + l16) * KM;
        v16bf bpq[4], bqf[4];
#pragma unroll
        for (int kt = 0; kt < 4; ++kt) {  // batch all 8 B frags, then 8 WMMAs
            bpq[kt] = load_frag(Bpq + kt * 32, half);
            bqf[kt] = load_frag(Bq  + kt * 32, half);
        }
        v8f X = (v8f)(0.f), Y = (v8f)(0.f);
#pragma unroll
        for (int kt = 0; kt < 4; ++kt) {
            X = wmma_bf16(apq[kt], bpq[kt], X);
            Y = wmma_bf16(aq [kt], bqf[kt], Y);
        }
#pragma unroll
        for (int v = 0; v < 8; ++v) {
            int a = v + half * 8;         // C layout: lanes>=16 hold M=v+8
            float x = X[v] / (Y[v] + 1e-6f);
            agg[r][h * 256 + a * 16 + l16] = f2bf(x);
        }
    }

    // ---- wait for the async x2d DMA, then concat into channels 1024..1151 ----
    asm volatile("s_wait_asynccnt 0" ::: "memory");
    __syncthreads();
    for (int e = tid; e < 16 * 128; e += 256) {
        int r = e >> 7, t = e & 127;
        agg[r][1024 + t] = f2bf(x2ds[r][t]);
    }
    __syncthreads();

    // ---- stage 2: output GEMM 16x256, K=1152 (bf16 WMMA from LDS + L2) ----
    v8f acc0 = (v8f)(0.f), acc1 = (v8f)(0.f);
    int nt0 = wave * 2, nt1 = wave * 2 + 1;
    const u16* B0 = wot + (size_t)(nt0 * 16 + l16) * PROJ_IN;
    const u16* B1 = wot + (size_t)(nt1 * 16 + l16) * PROJ_IN;
    const u16* Arow = &agg[l16][0];
    for (int kt = 0; kt < PROJ_IN / 32; ++kt) {
        int k0 = kt * 32;
        v16bf a  = load_frag(Arow + k0, half);
        v16bf b0 = load_frag(B0 + k0, half);
        v16bf b1 = load_frag(B1 + k0, half);
        acc0 = wmma_bf16(a, b0, acc0);
        acc1 = wmma_bf16(a, b1, acc1);
    }
    int c0 = nt0 * 16 + l16, c1 = nt1 * 16 + l16;
    float b0v = bo[c0], b1v = bo[c1];
#pragma unroll
    for (int v = 0; v < 8; ++v) {
        int r = v + half * 8;
        otile[r][c0] = acc0[v] + b0v;
        otile[r][c1] = acc1[v] + b1v;
    }
    __syncthreads();

    // ---- stage 3: LayerNorm over 256 ch + ReLU + store ----
#pragma unroll
    for (int rr = 0; rr < 2; ++rr) {
        int r = wave * 2 + rr;
        int cb = lane * 8;
        float vals[8], s = 0.f;
#pragma unroll
        for (int t = 0; t < 8; ++t) { vals[t] = otile[r][cb + t]; s += vals[t]; }
        float mean = wred32(s) * (1.f / 256.f);
        float vs = 0.f;
#pragma unroll
        for (int t = 0; t < 8; ++t) { float d = vals[t] - mean; vs += d * d; }
        float inv = rsqrtf(wred32(vs) * (1.f / 256.f) + 1e-5f);
        int j = jb * 16 + r;
        float* orow = out + ((size_t)i * LL + j) * AGG + cb;
        float o[8];
#pragma unroll
        for (int t = 0; t < 8; ++t) {
            int ch = cb + t;
            float y = (vals[t] - mean) * inv * go[ch] + betao[ch];
            o[t] = fmaxf(y, 0.f);
        }
        ((float4*)orow)[0] = make_float4(o[0], o[1], o[2], o[3]);
        ((float4*)orow)[1] = make_float4(o[4], o[5], o[6], o[7]);
    }
}

// ---------- host launch ----------
extern "C" void kernel_launch(void* const* d_in, const int* in_sizes, int n_in,
                              void* d_out, int out_size, void* d_ws, size_t ws_size,
                              hipStream_t stream) {
    const float* x1d   = (const float*)d_in[0];
    const float* x2d   = (const float*)d_in[1];
    const float* Wp    = (const float*)d_in[2];
    const float* bp    = (const float*)d_in[3];
    const float* gp    = (const float*)d_in[4];
    const float* betap = (const float*)d_in[5];
    const float* Wq    = (const float*)d_in[6];
    const float* bq    = (const float*)d_in[7];
    const float* Wo    = (const float*)d_in[8];
    const float* bo    = (const float*)d_in[9];
    const float* go    = (const float*)d_in[10];
    const float* betao = (const float*)d_in[11];
    float* out = (float*)d_out;

    char* ws = (char*)d_ws;
    u16*   x1dbf = (u16*)(ws);                          // 32768*384*2 = 25165824
    u16*   wt    = (u16*)(ws + 25165824);               // 128*384*2   =    98304
    u16*   wot   = (u16*)(ws + 25264128);               // 256*1152*2  =   589824
    float* Praw  = (float*)(ws + 25853952);             // 32768*64*4  =  8388608
    float* Qraw  = (float*)(ws + 34242560);             // 8388608
    float* cmax  = (float*)(ws + 42631168);             // 256*64*4    =    65536
    u16*   PQT   = (u16*)(ws + 42696704);               // 4*4096*128*2=  4194304
    u16*   QT    = (u16*)(ws + 46891008);               // 4194304  (end 51085312)

    k_cvt_x1d<<<6144, 256, 0, stream>>>(x1d, x1dbf);
    k_cvt_w  <<<192,  256, 0, stream>>>(Wp, Wq, wt);
    k_cvt_wo <<<1152, 256, 0, stream>>>(Wo, wot);
    k_gemm_pq<<<256,  256, 0, stream>>>(x1dbf, wt, bp, bq, Praw, Qraw);
    k_colmax <<<64,   256, 0, stream>>>(Qraw, cmax);
    k_lnexp  <<<4096, 256, 0, stream>>>(Praw, Qraw, cmax, gp, betap, PQT, QT);
    k_fused  <<<dim3(16, 256), 256, 0, stream>>>(PQT, QT, x2d, wot, bo, go, betao, out);
}
